// DecoderRNN_33732673143398
// MI455X (gfx1250) — matrix operand order repaired and checked by reading, not compiled, not run on hardware
//
#include <hip/hip_runtime.h>
#include <hip/hip_bf16.h>

typedef __attribute__((ext_vector_type(16))) __bf16 v16bf;
typedef __attribute__((ext_vector_type(8)))  float  v8f;

// Split workgroup barrier (CDNA5): make our LDS writes visible, then arrive;
// wait later so global stores / next-step loads overlap the barrier.
#define WG_BARRIER_ARRIVE() \
  asm volatile("s_wait_dscnt 0x0\ns_barrier_signal -1" ::: "memory")
#define WG_BARRIER_WAIT() asm volatile("s_barrier_wait -1" ::: "memory")

// ---------------------------------------------------------------------------
// Fast activations: gfx1250 has V_TANH_F32 (TRANS32). Sigmoid via tanh
// identity avoids the precise-division expansion of 1/(1+exp(-x)).
// ---------------------------------------------------------------------------
__device__ __forceinline__ float fast_tanh(float x) {
#if __has_builtin(__builtin_amdgcn_tanhf)
  return __builtin_amdgcn_tanhf(x);
#else
  const float e = __expf(2.0f * x);
  return 1.0f - 2.0f * __builtin_amdgcn_rcpf(e + 1.0f);
#endif
}

__device__ __forceinline__ float fast_sigmoid(float x) {
#if __has_builtin(__builtin_amdgcn_tanhf)
  return 0.5f + 0.5f * __builtin_amdgcn_tanhf(0.5f * x);
#else
  return __builtin_amdgcn_rcpf(1.0f + __expf(-x));
#endif
}

// ---------------------------------------------------------------------------
// WMMA fragment loaders (bf16 16x16x32, wave32 layouts per CDNA5 ISA 7.12.2)
// A 16x32: lane<16 -> M=lane, K in {0..7,16..23}; lane>=16 -> K in {8..15,24..31}
// B 32x16: lane holds column N=lane%16; K same as A: lane<16 {0..7,16..23}, lane>=16 {8..15,24..31}
// C/D 16x16 f32: VGPR v -> M = v (lanes 0-15) / v+8 (lanes 16-31), N = lane%16
// ---------------------------------------------------------------------------
template <typename SrcT>
__device__ __forceinline__ v16bf load_afrag(const SrcT* src, int ld, int m,
                                            int k0, int kb8) {
  v16bf f;
#pragma unroll
  for (int j = 0; j < 16; ++j) {
    const int k = k0 + ((j < 8) ? 0 : 16) + kb8 + (j & 7);
    f[j] = (__bf16)src[m * ld + k];
  }
  return f;
}

template <typename SrcT>
__device__ __forceinline__ v16bf load_bfrag(const SrcT* src, int ld, int n,
                                            int k0, int kb16) {
  v16bf f;
#pragma unroll
  for (int j = 0; j < 16; ++j) f[j] = (__bf16)src[n * ld + k0 + ((j < 8) ? 0 : 16) + (kb16 >> 1) + (j & 7)];
  return f;
}

// ---------------------------------------------------------------------------
// Input-projection GEMM: gx[rows x 384] = in[rows x K] * w_ih[384 x K]^T + bias
// Bias folded here (off the serial scan path): r/z gates get b_ih+b_hh,
// n gate gets b_ih only (its b_hh term lives inside r*(...) in the scan).
// ---------------------------------------------------------------------------
template <int K, bool BF16IN>
__global__ __launch_bounds__(256) void gemm_gx_kernel(
    const void* __restrict__ in_v, const float* __restrict__ w_ih,
    const float* __restrict__ b_ih, const float* __restrict__ b_hh,
    float* __restrict__ gx) {
  const int tid  = threadIdx.x;
  const int lane = tid & 31;
  const int wv   = tid >> 5;
  const int col  = lane & 15;
  const int kb8  = (lane >> 4) << 3;
  const int kb16 = (lane >> 4) << 4;

  const int tile = blockIdx.x * 8 + wv;
  const int mt   = tile / 24;
  const int nt   = tile % 24;

  const int m = mt * 16 + col;
  const int n = nt * 16 + col;

  v8f acc;
#pragma unroll
  for (int v = 0; v < 8; ++v) acc[v] = 0.0f;

#pragma unroll
  for (int c = 0; c < (K >> 5); ++c) {
    v16bf a;
    if (BF16IN) {
      a = load_afrag<__bf16>((const __bf16*)in_v, K, m, c * 32, kb8);
    } else {
      a = load_afrag<float>((const float*)in_v, K, m, c * 32, kb8);
    }
    const v16bf b = load_bfrag<float>(w_ih, K, n, c * 32, kb16);
    acc = __builtin_amdgcn_wmma_f32_16x16x32_bf16(false, a, false, b,
                                                  (short)0, acc, false, false);
  }

  const float bias = b_ih[n] + ((nt < 16) ? b_hh[n] : 0.0f);

#pragma unroll
  for (int v = 0; v < 8; ++v) {
    const int M = mt * 16 + ((lane < 16) ? v : v + 8);
    gx[M * 384 + nt * 16 + col] = acc[v] + bias;
  }
}

// ---------------------------------------------------------------------------
// One GRU time step (inlined twice for ping-pong register pipelining).
// Consumes gx regs (gr,gz,gn) for step t; issues loads for t+1 into
// (grn,gzn,gnn) — a full WMMA+barrier window of latency cover, so no
// explicit prefetch is needed. h state carried in hprev registers; LDS
// tile only feeds the WMMA A-fragments. Split barrier: arrive right
// after LDS writes, wait at the top of the next step.
// ---------------------------------------------------------------------------
__device__ __forceinline__ void gru_step(
    const __bf16* __restrict__ hr, __bf16* __restrict__ hw,
    const float* __restrict__ gx, float (&gr)[8], float (&gz)[8],
    float (&gn)[8], float (&grn)[8], float (&gzn)[8], float (&gnn)[8],
    const v16bf (&Br)[4], const v16bf (&Bz)[4], const v16bf (&Bn)[4],
    float bhh_n, float (&hprev)[8], int (&gxoff)[8], int (&hoff)[8],
    __bf16* __restrict__ h_out, int col, int kb8, const int (&Mv)[8],
    int jcol) {
  WG_BARRIER_WAIT();  // writes of previous step now visible

  // A fragments of current h tile
  v16bf A[4];
#pragma unroll
  for (int c = 0; c < 4; ++c)
    A[c] = load_afrag<__bf16>(hr, 128, col, c * 32, kb8);

  // issue next step's gx loads (consumed after the next barrier)
#pragma unroll
  for (int v = 0; v < 8; ++v) {
    grn[v] = gx[gxoff[v]];
    gzn[v] = gx[gxoff[v] + 128];
    gnn[v] = gx[gxoff[v] + 256];
    gxoff[v] += 384;
  }

  v8f accr, accz, accn;
#pragma unroll
  for (int v = 0; v < 8; ++v) {
    accr[v] = 0.0f;
    accz[v] = 0.0f;
    accn[v] = bhh_n;
  }
#pragma unroll
  for (int c = 0; c < 4; ++c) {
    accr = __builtin_amdgcn_wmma_f32_16x16x32_bf16(false, A[c], false, Br[c],
                                                   (short)0, accr, false, false);
    accz = __builtin_amdgcn_wmma_f32_16x16x32_bf16(false, A[c], false, Bz[c],
                                                   (short)0, accz, false, false);
    accn = __builtin_amdgcn_wmma_f32_16x16x32_bf16(false, A[c], false, Bn[c],
                                                   (short)0, accn, false, false);
  }

  __bf16 hb8[8];
#pragma unroll
  for (int v = 0; v < 8; ++v) {
    const float r = fast_sigmoid(gr[v] + accr[v]);
    const float z = fast_sigmoid(gz[v] + accz[v]);
    const float n = fast_tanh(gn[v] + r * accn[v]);
    const float hnew = n + z * (hprev[v] - n);  // == (1-z)*n + z*h_old
    hprev[v] = hnew;
    hb8[v]   = (__bf16)hnew;
    hw[Mv[v] * 128 + jcol] = hb8[v];
  }

  WG_BARRIER_ARRIVE();  // LDS writes visible; others may proceed

  // global h trace stores overlap the barrier
#pragma unroll
  for (int v = 0; v < 8; ++v) {
    h_out[hoff[v]] = hb8[v];
    hoff[v] += 128;
  }
}

// ---------------------------------------------------------------------------
// Recurrent GRU scan. One block per 16-row batch tile; 8 waves.
// W_hh lives in registers as bf16 B-fragments (reused all 512 steps);
// h tile double-buffered in LDS; one split barrier per step.
// ---------------------------------------------------------------------------
__global__ __launch_bounds__(256) void gru_scan_kernel(
    const float* __restrict__ gx, const float* __restrict__ w_hh,
    const float* __restrict__ b_hh, __bf16* __restrict__ h_out,
    float* __restrict__ h_last, int T) {
  __shared__ __bf16 lds_h[2][16 * 128];

  const int tid   = threadIdx.x;
  const int lane  = tid & 31;
  const int wv    = tid >> 5;
  const int col   = lane & 15;
  const int kb8   = (lane >> 4) << 3;
  const int kb16  = (lane >> 4) << 4;
  const int bbase = blockIdx.x * 16;
  const int jcol  = wv * 16 + col;

  // h0 = 0 (both buffers)
  for (int i = tid; i < 2 * 16 * 128; i += 256)
    lds_h[0][i] = (__bf16)0.0f;

  // W_hh -> bf16 B fragments (persist in registers for all T steps)
  v16bf Br[4], Bz[4], Bn[4];
#pragma unroll
  for (int c = 0; c < 4; ++c) {
    Br[c] = load_bfrag<float>(w_hh, 128, 0 * 128 + jcol, c * 32, kb16);
    Bz[c] = load_bfrag<float>(w_hh, 128, 1 * 128 + jcol, c * 32, kb16);
    Bn[c] = load_bfrag<float>(w_hh, 128, 2 * 128 + jcol, c * 32, kb16);
  }
  const float bhh_n = b_hh[256 + jcol];

  int Mv[8], gxoff[8], hoff[8];
  float hprev[8];
#pragma unroll
  for (int v = 0; v < 8; ++v) {
    Mv[v]    = (lane < 16) ? v : v + 8;
    const int row0 = (bbase + Mv[v]) * T;
    gxoff[v] = row0 * 384 + jcol;
    hoff[v]  = row0 * 128 + jcol;
    hprev[v] = 0.0f;
  }

  // preload gx for t=0 (pipeline prologue)
  float grA[8], gzA[8], gnA[8], grB[8], gzB[8], gnB[8];
#pragma unroll
  for (int v = 0; v < 8; ++v) {
    grA[v] = gx[gxoff[v]];
    gzA[v] = gx[gxoff[v] + 128];
    gnA[v] = gx[gxoff[v] + 256];
    gxoff[v] += 384;
  }

  WG_BARRIER_ARRIVE();  // zero-init visible; pairs with first step's wait

  for (int t = 0; t < T; t += 2) {
    gru_step(lds_h[0], lds_h[1], gx, grA, gzA, gnA, grB, gzB, gnB, Br, Bz, Bn,
             bhh_n, hprev, gxoff, hoff, h_out, col, kb8, Mv, jcol);
    gru_step(lds_h[1], lds_h[0], gx, grB, gzB, gnB, grA, gzA, gnA, Br, Bz, Bn,
             bhh_n, hprev, gxoff, hoff, h_out, col, kb8, Mv, jcol);
  }

  // final hidden state (f32, from registers)
#pragma unroll
  for (int v = 0; v < 8; ++v)
    h_last[(bbase + Mv[v]) * 128 + jcol] = hprev[v];
}

// ---------------------------------------------------------------------------
// FC head: out[256 x 6] = h_last[256 x 128] * fc_w[6 x 128]^T + fc_b
// ---------------------------------------------------------------------------
__global__ __launch_bounds__(256) void fc_kernel(
    const float* __restrict__ h_last, const float* __restrict__ fc_w,
    const float* __restrict__ fc_b, float* __restrict__ out) {
  const int idx = blockIdx.x * blockDim.x + threadIdx.x;
  if (idx >= 256 * 6) return;
  const int b = idx / 6, c = idx % 6;
  float s = fc_b[c];
#pragma unroll 4
  for (int k = 0; k < 128; ++k) s += h_last[b * 128 + k] * fc_w[c * 128 + k];
  out[idx] = s;
}

// ---------------------------------------------------------------------------
extern "C" void kernel_launch(void* const* d_in, const int* in_sizes, int n_in,
                              void* d_out, int out_size, void* d_ws,
                              size_t ws_size, hipStream_t stream) {
  const float* x     = (const float*)d_in[0];  // (256,512,64)
  const float* w_ih0 = (const float*)d_in[1];  // (384,64)
  const float* w_ih1 = (const float*)d_in[2];  // (384,128)
  const float* w_ih2 = (const float*)d_in[3];  // (384,128)
  const float* w_hh  = (const float*)d_in[4];  // (3,384,128)
  const float* b_ih  = (const float*)d_in[5];  // (3,384)
  const float* b_hh  = (const float*)d_in[6];  // (3,384)
  const float* fc_w  = (const float*)d_in[7];  // (6,128)
  const float* fc_b  = (const float*)d_in[8];  // (6,)

  const int Bsz = 256, T = 512, ROWS = Bsz * T;

  char* ws = (char*)d_ws;
  float*  gx = (float*)ws;                                   // 192 MB, reused
  __bf16* hA = (__bf16*)(ws + (size_t)ROWS * 384 * 4);       // 32 MB
  __bf16* hB = (__bf16*)(ws + (size_t)ROWS * 384 * 4 + (size_t)ROWS * 128 * 2);
  float*  h_last =
      (float*)(ws + (size_t)ROWS * 384 * 4 + 2 * (size_t)ROWS * 128 * 2);

  const dim3 gblk(256);
  const dim3 ggrd((ROWS / 16) * 24 / 8);
  const dim3 sgrd(Bsz / 16);

  // Layer 0
  gemm_gx_kernel<64, false><<<ggrd, gblk, 0, stream>>>(x, w_ih0, b_ih, b_hh, gx);
  gru_scan_kernel<<<sgrd, gblk, 0, stream>>>(gx, w_hh, b_hh, hA, h_last, T);
  // Layer 1
  gemm_gx_kernel<128, true><<<ggrd, gblk, 0, stream>>>(hA, w_ih1, b_ih + 384,
                                                       b_hh + 384, gx);
  gru_scan_kernel<<<sgrd, gblk, 0, stream>>>(gx, w_hh + 384 * 128, b_hh + 384,
                                             hB, h_last, T);
  // Layer 2
  gemm_gx_kernel<128, true><<<ggrd, gblk, 0, stream>>>(hB, w_ih2, b_ih + 2 * 384,
                                                       b_hh + 2 * 384, gx);
  gru_scan_kernel<<<sgrd, gblk, 0, stream>>>(gx, w_hh + 2 * 384 * 128,
                                             b_hh + 2 * 384, hA, h_last, T);
  // FC head
  fc_kernel<<<dim3(6), gblk, 0, stream>>>(h_last, fc_w, fc_b, (float*)d_out);
}
